// SelfAttention_72327249265164
// MI455X (gfx1250) — matrix operand order repaired and checked by reading, not compiled, not run on hardware
//
#include <hip/hip_runtime.h>
#include <stdint.h>

typedef __attribute__((ext_vector_type(16))) _Float16 v16h;
typedef __attribute__((ext_vector_type(8)))  float    v8f;

#define B_   64
#define M_   2048
#define H_   64
#define BLKF 256.0f   // reference block_size
#define KT   32       // keys per WMMA step (K of 16x16x32)
#define PAD  34       // padded K-stride of LDS tile, in f16 units (keeps 4B align)

__global__ __launch_bounds__(256)
void autoselect_attn_kernel(const float* __restrict__ span,
                            const float* __restrict__ value,
                            float* __restrict__ out)
{
    const int tid   = threadIdx.x;
    const int lane  = tid & 31;
    const int wv    = tid >> 5;              // wave in workgroup, 0..7
    const int b     = blockIdx.x >> 4;       // batch
    const int qblk  = (blockIdx.x & 15) * 128;
    const int qbase = qblk + wv * 16;        // this wave's 16 queries

    const int m     = lane & 15;             // query row within tile
    const int khalf = lane >> 4;             // which K-half this lane holds (A frag)
    const int qrow  = qbase + m;

    // ---- per-row analytic window parameters ----
    const float s0 = span[((size_t)b * M_ + qrow) * 2 + 0];
    const float s1 = span[((size_t)b * M_ + qrow) * 2 + 1];
    const float mu = (float)qrow + s0 * BLKF;
    const float sp = (s1 > 20.0f) ? s1 : log1pf(__expf(s1));   // softplus
    const float sigma = sp * BLKF + 1e-3f;
    // score(j) = -(j-mu)^2 / (2 sigma^2) / sqrt(H),  sqrt(64) = 8
    const float cns = 1.0f / (16.0f * sigma * sigma);
    // analytic row max: nearest valid integer key to mu
    const float jstar = fminf(fmaxf(roundf(mu), 0.0f), (float)(M_ - 1));
    const float dstar = jstar - mu;
    const float smax  = -(dstar * dstar) * cns;

    // V tile staged transposed: tile[h][k], f16, padded rows
    __shared__ _Float16 tile[H_ * PAD];

    v8f acc0 = {}, acc1 = {}, acc2 = {}, acc3 = {};
    float rowsum = 0.0f;

    // staging decomposition: 256 threads -> (key k, 8-wide h group)
    const int sk  = tid >> 3;        // 0..31 : key within block
    const int shg = (tid & 7) * 8;   // h0 of the 8 h's this thread converts

    const int n = lane & 15;         // B/C fragment column this lane owns

    for (int kb = 0; kb < M_ / KT; ++kb) {
        const int j0 = kb * KT;

        __syncthreads();  // prior iteration's tile reads are done
        {   // load a 32x64 f32 V tile, convert to f16, store transposed
            const float4* vsrc =
                (const float4*)(value + ((size_t)b * M_ + (j0 + sk)) * H_ + shg);
            float4 va = vsrc[0];
            float4 vb = vsrc[1];
            _Float16 hv[8] = {
                (_Float16)va.x, (_Float16)va.y, (_Float16)va.z, (_Float16)va.w,
                (_Float16)vb.x, (_Float16)vb.y, (_Float16)vb.z, (_Float16)vb.w };
            #pragma unroll
            for (int i = 0; i < 8; ++i)
                tile[(shg + i) * PAD + sk] = hv[i];
        }
        __syncthreads();  // tile visible to all waves

        // ---- A fragment: un-normalized softmax weights, f16 ----
        // A 16x32 layout (matching B): lane(m,khalf), VGPR v half hh holds
        //   k = khalf*16 + v*2 + hh
        v16h a;
        #pragma unroll
        for (int v = 0; v < 8; ++v) {
            #pragma unroll
            for (int hh = 0; hh < 2; ++hh) {
                const int k = (khalf << 4) + (v << 1) + hh;
                const float d = (float)(j0 + k) - mu;
                const float e = __expf(-(d * d) * cns - smax);
                rowsum += e;
                a[v * 2 + hh] = (_Float16)e;
            }
        }

        // ---- B fragments from LDS + 4 WMMAs (H = 4 x 16) ----
        // B 32x16 layout used here (same k order as A): lane (n + 16*kh) holds col n,
        //   K = kh*16 + 2v + {0,1} in VGPR v lo/hi -> contiguous f16 pair in tile
        const uint32_t* t32 = (const uint32_t*)tile;
        #define DO_NTILE(T, ACC)                                                   \
        {                                                                          \
            union { uint32_t u[8]; v16h h; } bf;                                   \
            const int rbase = (((T) * 16 + n) * PAD + khalf * 16) >> 1;            \
            _Pragma("unroll")                                                      \
            for (int v = 0; v < 8; ++v) bf.u[v] = t32[rbase + v];                  \
            ACC = __builtin_amdgcn_wmma_f32_16x16x32_f16(                          \
                false, a, false, bf.h, (short)0, ACC, false, false);               \
        }
        DO_NTILE(0, acc0)
        DO_NTILE(1, acc1)
        DO_NTILE(2, acc2)
        DO_NTILE(3, acc3)
        #undef DO_NTILE
    }

    // ---- finish softmax normalization ----
    // lane L holds partial sum over its 16 k's per block; partner lane L^16
    // holds the other 16 -> one xor-shuffle completes the row sum.
    const float other = __shfl_xor(rowsum, 16, 32);
    const float inv   = 1.0f / (rowsum + other);   // valid for row m = lane&15

    // C/D 16x16 f32 layout: lane L -> col n = L&15, VGPR r -> row r + 8*(L>=16)
    #pragma unroll
    for (int r = 0; r < 8; ++r) {
        const int row   = khalf * 8 + r;
        const float ivr = __shfl(inv, row, 32);    // lane 'row' owns that row's sum
        const size_t ob = ((size_t)b * M_ + (qbase + row)) * H_ + n;
        out[ob +  0] = acc0[r] * ivr;
        out[ob + 16] = acc1[r] * ivr;
        out[ob + 32] = acc2[r] * ivr;
        out[ob + 48] = acc3[r] * ivr;
    }
}

extern "C" void kernel_launch(void* const* d_in, const int* in_sizes, int n_in,
                              void* d_out, int out_size, void* d_ws, size_t ws_size,
                              hipStream_t stream) {
    (void)in_sizes; (void)n_in; (void)out_size; (void)d_ws; (void)ws_size;
    const float* span  = (const float*)d_in[0];
    const float* value = (const float*)d_in[1];
    float*       out   = (float*)d_out;

    // grid: 64 batches x (2048 queries / 128 per workgroup) = 1024 workgroups
    dim3 grid(B_ * (M_ / 128));
    dim3 block(256);
    autoselect_attn_kernel<<<grid, block, 0, stream>>>(span, value, out);
}
